// MultiHeadAttention_46102178955366
// MI455X (gfx1250) — matrix operand order repaired and checked by reading, not compiled, not run on hardware
//
#include <hip/hip_runtime.h>

typedef __attribute__((ext_vector_type(16))) _Float16 v16h;
typedef __attribute__((ext_vector_type(8)))  _Float16 v8h;
typedef __attribute__((ext_vector_type(8)))  float    v8f;

constexpr int BB = 8, NN = 2048, HH = 4, DD = 64, OUTD = 256;

__device__ __forceinline__ v8f wmma16(v16h a, v16h b, v8f c) {
    // D = A(16x32 f16) * B(32x16 f16) + C(16x16 f32)
    return __builtin_amdgcn_wmma_f32_16x16x32_f16(false, a, false, b, (short)0, c, false, false);
}

// ---------------------------------------------------------------------------
// Kernel 1: per-head QKV projection.  Q/K/V[b,h,n,d] (f16) = x_h @ W + bias
// Block = 256 threads = 8 waves; all waves in a block share (b,h).
// ---------------------------------------------------------------------------
__global__ __launch_bounds__(256) void qkv_proj(
    const float* __restrict__ x,
    const float* __restrict__ Wq, const float* __restrict__ Wk, const float* __restrict__ Wv,
    const float* __restrict__ bq, const float* __restrict__ bk, const float* __restrict__ bv,
    _Float16* __restrict__ Qh, _Float16* __restrict__ Kh, _Float16* __restrict__ Vh)
{
    __shared__ _Float16 WT[3][64][64];   // [mat][n][k] f16 (transposed for contiguous B-frags)

    const int tid  = threadIdx.x;
    const int w    = tid >> 5;
    const int lane = tid & 31;
    const int g    = lane >> 4;          // lane half (0/1)
    const int qn   = lane & 15;          // column / row-in-tile index
    const int gw   = blockIdx.x * 8 + w;
    const int mt   = gw & 127;           // 16-row tile index within N
    const int h    = (gw >> 7) & 3;
    const int b    = gw >> 9;

    const float* Wsrc[3] = {Wq, Wk, Wv};
    for (int idx = tid; idx < 3 * 4096; idx += 256) {
        int mat = idx >> 12, r = idx & 4095, k = r >> 6, n = r & 63;
        WT[mat][n][k] = (_Float16)Wsrc[mat][h * 4096 + r];
    }
    __syncthreads();

    // A fragments of x_h (16 rows x 64 k), converted f32->f16
    const float* xb = x + ((size_t)b * NN + (size_t)mt * 16 + qn) * (HH * DD) + h * DD;
    v16h A2[2];
#pragma unroll
    for (int kc = 0; kc < 2; kc++) {
        const float* p = xb + 32 * kc + 16 * g;
        v16h a;
#pragma unroll
        for (int i = 0; i < 8; i++) { a[i] = (_Float16)p[i]; a[8 + i] = (_Float16)p[8 + i]; }
        A2[kc] = a;
    }

    const float* bias3[3] = {bq, bk, bv};
    _Float16*    out3[3]  = {Qh, Kh, Vh};
#pragma unroll
    for (int mat = 0; mat < 3; mat++) {
#pragma unroll
        for (int nt = 0; nt < 4; nt++) {
            float bias = bias3[mat][h * DD + nt * 16 + qn];
            v8f acc;
#pragma unroll
            for (int i = 0; i < 8; i++) acc[i] = bias;
#pragma unroll
            for (int kc = 0; kc < 2; kc++) {
                v16h Bf = *(const v16h*)&WT[mat][nt * 16 + qn][32 * kc + 16 * g];
                acc = wmma16(A2[kc], Bf, acc);
            }
            _Float16* op = out3[mat] + ((size_t)(b * HH + h) * NN + (size_t)mt * 16) * DD + nt * 16 + qn;
#pragma unroll
            for (int r = 0; r < 8; r++) op[(size_t)(r + 8 * g) * DD] = (_Float16)acc[r];
        }
    }
}

// ---------------------------------------------------------------------------
// Kernel 2: flash attention, fully transposed.  S^T = K @ Q^T so softmax
// stats are per-lane; O^T = V^T @ P^T so rescale/divide need no shuffles and
// the epilogue is 4x global_store_b128.  P^T B-fragment built in registers
// via cvt_pk + one shfl_xor(16).  K tile staged with async global->LDS DMA.
// ---------------------------------------------------------------------------
__global__ __launch_bounds__(256) void attn_flash(
    const _Float16* __restrict__ Qh, const _Float16* __restrict__ Kh,
    const _Float16* __restrict__ Vh, _Float16* __restrict__ Oh)
{
    __shared__ _Float16 Kt[32 * 64];     // [key][dim]
    __shared__ _Float16 Vt[64 * 32];     // [dim][key]  (transposed)

    const int tid  = threadIdx.x;
    const int lane = tid & 31;
    const int w    = tid >> 5;
    const int g    = lane >> 4;
    const int qn   = lane & 15;

    const int qb = blockIdx.x & 15;
    const int h  = (blockIdx.x >> 4) & 3;
    const int b  = blockIdx.x >> 6;
    const int qbase = qb * 128 + w * 16;
    const size_t bh = (size_t)(b * HH + h) * NN * DD;

    // Q as B-fragments (column = query qn, K = head dims): contiguous 16 halves
    v16h qB[2];
#pragma unroll
    for (int kc = 0; kc < 2; kc++)
        qB[kc] = *(const v16h*)(Qh + bh + (size_t)(qbase + qn) * DD + 32 * kc + 16 * g);

    float m = -__builtin_inff();
    float l = 0.0f;
    v8f acc[4];                          // O^T tiles: rows = dims 16t+r+8g, col = query qn
#pragma unroll
    for (int t = 0; t < 4; t++)
#pragma unroll
        for (int r = 0; r < 8; r++) acc[t][r] = 0.0f;

    const int row = tid >> 3;            // key row staged by this thread
    const int off = (tid & 7) * 8;       // dim offset (8 halves)

    for (int kb = 0; kb < NN / 32; kb++) {
        __syncthreads();
        {   // K tile: async DMA global -> LDS (row-major copy, b128/lane)
            const _Float16* kg = Kh + bh + (size_t)(kb * 32 + row) * DD + off;
            unsigned laddr = (unsigned)(size_t)&Kt[row * 64 + off];
            asm volatile("global_load_async_to_lds_b128 %0, %1, off"
                         :: "v"(laddr), "v"(kg) : "memory");
            // V tile: staged transposed [dim][key]
            v8h vv = *(const v8h*)(Vh + bh + (size_t)(kb * 32 + row) * DD + off);
#pragma unroll
            for (int j = 0; j < 8; j++) Vt[(off + j) * 32 + row] = vv[j];
            asm volatile("s_wait_asynccnt 0" ::: "memory");
        }
        __syncthreads();

        // S^T tiles: rows = keys (16n + r + 8g), col = query qn
        v8f s[2];
#pragma unroll
        for (int n = 0; n < 2; n++) {
            const _Float16* ap = &Kt[(16 * n + qn) * 64];
#pragma unroll
            for (int r = 0; r < 8; r++) s[n][r] = 0.0f;
#pragma unroll
            for (int kc = 0; kc < 2; kc++) {
                v16h aK;
#pragma unroll
                for (int i = 0; i < 8; i++) {
                    aK[i]     = ap[32 * kc + 16 * g + i];
                    aK[8 + i] = ap[32 * kc + 16 * g + 8 + i];
                }
                s[n] = wmma16(aK, qB[kc], s[n]);
            }
#pragma unroll
            for (int r = 0; r < 8; r++) s[n][r] *= 0.125f;  // 1/sqrt(64)
        }

        // online softmax along keys (per-lane query column)
        float vmax = s[0][0];
#pragma unroll
        for (int n = 0; n < 2; n++)
#pragma unroll
            for (int r = 0; r < 8; r++) vmax = fmaxf(vmax, s[n][r]);
        vmax = fmaxf(vmax, __shfl_xor(vmax, 16, 32));
        float mnew = fmaxf(m, vmax);
        float corr = __expf(m - mnew);
        float lsum = 0.0f;
        v8f p[2];
#pragma unroll
        for (int n = 0; n < 2; n++)
#pragma unroll
            for (int r = 0; r < 8; r++) { p[n][r] = __expf(s[n][r] - mnew); lsum += p[n][r]; }
        lsum += __shfl_xor(lsum, 16, 32);
        l = l * corr + lsum;
        m = mnew;

        // rescale accumulators: corr is per-lane (per query column) -> no shuffles
#pragma unroll
        for (int t = 0; t < 4; t++)
#pragma unroll
            for (int r = 0; r < 8; r++) acc[t][r] *= corr;

        // Build P^T B-fragment in registers:
        // lane (qn,g) needs keys 16g..16g+15 of column qn.  Own tile-g regs
        // hold keys 16g+8g..; partner lane (qn^16) tile-g regs hold the rest.
        unsigned pk0[4], pk1[4];
#pragma unroll
        for (int j = 0; j < 4; j++) {
            union { _Float16 hh[2]; unsigned u; } c0, c1;
            c0.hh[0] = (_Float16)p[0][2 * j]; c0.hh[1] = (_Float16)p[0][2 * j + 1];
            c1.hh[0] = (_Float16)p[1][2 * j]; c1.hh[1] = (_Float16)p[1][2 * j + 1];
            pk0[j] = c0.u; pk1[j] = c1.u;
        }
        union { v16h v; unsigned u[8]; } Bp;
#pragma unroll
        for (int j = 0; j < 4; j++) {
            unsigned own  = g ? pk1[j] : pk0[j];           // my tile-g pack
            unsigned send = g ? pk0[j] : pk1[j];           // partner wants tile (1-g)... of itself
            unsigned oth  = (unsigned)__shfl_xor((int)send, 16, 32); // partner's tile-g pack
            Bp.u[j]     = g ? oth : own;                   // keys 16g + 0..7
            Bp.u[4 + j] = g ? own : oth;                   // keys 16g + 8..15
        }

        // O^T += V^T(16 dims x 32 keys) @ P^T(32 keys x 16 queries)
#pragma unroll
        for (int t = 0; t < 4; t++) {
            const _Float16* vp = &Vt[(16 * t + qn) * 32];
            v16h aV;
#pragma unroll
            for (int i = 0; i < 8; i++) { aV[i] = vp[16 * g + i]; aV[8 + i] = vp[16 * g + 8 + i]; }
            acc[t] = wmma16(aV, Bp.v, acc[t]);
        }
    }

    // epilogue: per-lane divide, contiguous b128 stores (8 dims per store)
    float linv = 1.0f / l;
#pragma unroll
    for (int t = 0; t < 4; t++) {
        v8h ov;
#pragma unroll
        for (int r = 0; r < 8; r++) ov[r] = (_Float16)(acc[t][r] * linv);
        *(v8h*)(Oh + bh + (size_t)(qbase + qn) * DD + 16 * t + 8 * g) = ov;
    }
}

// ---------------------------------------------------------------------------
// Kernel 3: output projection  out[B*N,256] = concat(O) @ Wp + bp  (f32 out)
// Each block owns a 128-row x 64-col tile; Wp strip staged transposed in LDS.
// ---------------------------------------------------------------------------
__global__ __launch_bounds__(256) void out_proj(
    const _Float16* __restrict__ Oh, const float* __restrict__ Wp,
    const float* __restrict__ bp, float* __restrict__ out)
{
    __shared__ _Float16 WT[64 * 256];    // [col][k] f16

    const int tid  = threadIdx.x;
    const int w    = tid >> 5;
    const int lane = tid & 31;
    const int g    = lane >> 4;
    const int qn   = lane & 15;
    const int cg   = blockIdx.x & 3;
    const int rg   = blockIdx.x >> 2;
    const int colbase = cg * 64;
    const int mbase   = rg * 128 + w * 16;

    for (int idx = tid; idx < 64 * 256; idx += 256) {
        int c = idx & 63, k = idx >> 6;
        WT[c * 256 + k] = (_Float16)Wp[k * 256 + colbase + c];
    }
    __syncthreads();

    const int rr = mbase + qn;
    const int b  = rr >> 11;
    const int n  = rr & 2047;

    v8f acc[4];
#pragma unroll
    for (int nt = 0; nt < 4; nt++) {
        float bias = bp[colbase + nt * 16 + qn];
#pragma unroll
        for (int r = 0; r < 8; r++) acc[nt][r] = bias;
    }

#pragma unroll
    for (int kc = 0; kc < 8; kc++) {
        v16h aO;
        int col0 = 32 * kc + 16 * g;
        int col1 = 32 * kc + 16 * g + 8;
        const _Float16* p0 = Oh + (((size_t)b * HH + (col0 >> 6)) * NN + n) * DD + (col0 & 63);
        const _Float16* p1 = Oh + (((size_t)b * HH + (col1 >> 6)) * NN + n) * DD + (col1 & 63);
#pragma unroll
        for (int i = 0; i < 8; i++) { aO[i] = p0[i]; aO[8 + i] = p1[i]; }
#pragma unroll
        for (int nt = 0; nt < 4; nt++) {
            v16h bW = *(const v16h*)&WT[(nt * 16 + qn) * 256 + 32 * kc + 16 * g];
            acc[nt] = wmma16(aO, bW, acc[nt]);
        }
    }

#pragma unroll
    for (int nt = 0; nt < 4; nt++) {
        float* op = out + (size_t)mbase * OUTD + colbase + nt * 16 + qn;
#pragma unroll
        for (int r = 0; r < 8; r++) op[(size_t)(r + 8 * g) * OUTD] = acc[nt][r];
    }
}

// ---------------------------------------------------------------------------
extern "C" void kernel_launch(void* const* d_in, const int* in_sizes, int n_in,
                              void* d_out, int out_size, void* d_ws, size_t ws_size,
                              hipStream_t stream) {
    const float* x  = (const float*)d_in[0];
    const float* Wq = (const float*)d_in[1];
    const float* Wk = (const float*)d_in[2];
    const float* Wv = (const float*)d_in[3];
    const float* bq = (const float*)d_in[4];
    const float* bk = (const float*)d_in[5];
    const float* bv = (const float*)d_in[6];
    const float* Wp = (const float*)d_in[7];
    const float* bp = (const float*)d_in[8];
    float* out = (float*)d_out;

    const size_t SZ = (size_t)BB * HH * NN * DD;   // 4M elems per tensor
    _Float16* Qh = (_Float16*)d_ws;
    _Float16* Kh = Qh + SZ;
    _Float16* Vh = Kh + SZ;
    _Float16* Oh = Vh + SZ;                        // total 32 MB f16 workspace

    qkv_proj<<<BB * HH * (NN / 16) / 8, 256, 0, stream>>>(x, Wq, Wk, Wv, bq, bk, bv, Qh, Kh, Vh);
    attn_flash<<<BB * HH * (NN / 128), 256, 0, stream>>>(Qh, Kh, Vh, Oh);
    out_proj<<<4 * (BB * NN / 128), 256, 0, stream>>>(Oh, Wp, bp, out);
}
